// TotalAttention_20512763806162
// MI455X (gfx1250) — matrix operand order repaired and checked by reading, not compiled, not run on hardware
//
#include <hip/hip_runtime.h>

#define HEADS 8
#define HEAD_DIM 32
#define NTOK 64
#define DIM 256
#define QKV_DIM 768

// LDS strides (in bf16 elements) -- keep 16B alignment of fragment chunks
#define XS  264   // x / attn_out tile stride
#define QKS 520   // q|k tile stride (q cols 0..255, k cols 256..511)
#define VTS 72    // v-transposed stride [256 dims][64 tokens]
#define PBS 72    // softmax P stride per wave [16 rows]

#define XB_OFF 0
#define QK_OFF 16896          // 64*XS
#define VT_OFF 50176          // QK_OFF + 64*QKS
#define PB_OFF 68608          // VT_OFF + 256*VTS
#define SMEM_ELEMS 77824      // PB_OFF + 8*16*PBS

typedef __attribute__((ext_vector_type(16))) __bf16 bf16x16;
typedef __attribute__((ext_vector_type(8)))  float  f32x8;

union BF16Frag { bf16x16 v; uint4 q[2]; };

__device__ __forceinline__ unsigned short f2bf(float f) {
    unsigned int u = __float_as_uint(f);
    u += 0x7FFFu + ((u >> 16) & 1u);   // round-to-nearest-even
    return (unsigned short)(u >> 16);
}

__device__ __forceinline__ f32x8 wmma_bf16(bf16x16 a, bf16x16 b, f32x8 c) {
    return __builtin_amdgcn_wmma_f32_16x16x32_bf16(
        false, a, false, b, (short)0, c, false, false);
}

// A-fragment (16xK tile, row-major, 16-bit): lane<16: row=lane, K elems {0..7,16..23};
// lane>=16: row=lane-16, K elems {8..15,24..31}  (ISA 7.12.2)
__device__ __forceinline__ bf16x16 load_a16(const unsigned short* p, int stride,
                                            int kstart, int lane) {
    int half = lane >> 4, lr = lane & 15;
    const unsigned short* base = p + lr * stride + kstart + half * 8;
    BF16Frag f;
    f.q[0] = *(const uint4*)(base);
    f.q[1] = *(const uint4*)(base + 16);
    return f.v;
}

// B-fragment from transposed ([N][K] row-major) source: lane<16: N=lane, K elems {0..7,16..23};
// lane>=16: N=lane-16, K elems {8..15,24..31}  (matches A-fragment)
__device__ __forceinline__ bf16x16 load_b16(const unsigned short* p, int stride,
                                            int kstart, int lane) {
    int half = lane >> 4, lr = lane & 15;
    const unsigned short* base = p + lr * stride + kstart + half * 8;
    BF16Frag f;
    f.q[0] = *(const uint4*)(base);
    f.q[1] = *(const uint4*)(base + 16);
    return f.v;
}

// One-time fp32 -> bf16 weight conversion + transpose into workspace:
// wt  : [768 cols][256 k]   (wq cols 0..255, wkv cols 256..767)
// wpt : [256 cols][256 k]
__global__ __launch_bounds__(256) void convert_weights_kernel(
    const float* __restrict__ wq, const float* __restrict__ wkv,
    const float* __restrict__ wp, unsigned short* __restrict__ wt,
    unsigned short* __restrict__ wpt) {
    int idx = blockIdx.x * 256 + threadIdx.x;
    const int T1 = QKV_DIM * DIM;
    if (idx < T1) {
        int c = idx / DIM, k = idx - c * DIM;
        float v = (c < DIM) ? wq[k * DIM + c] : wkv[k * (2 * DIM) + (c - DIM)];
        wt[idx] = f2bf(v);
    } else {
        int j = idx - T1;
        if (j < DIM * DIM) {
            int c = j / DIM, k = j - c * DIM;
            wpt[j] = f2bf(wp[k * DIM + c]);
        }
    }
}

__global__ __launch_bounds__(256) void TotalAttention_kernel(
    const float* __restrict__ x, const unsigned short* __restrict__ wt,
    const float* __restrict__ bq, const float* __restrict__ bkv,
    const unsigned short* __restrict__ wpt, const float* __restrict__ bp,
    const float* __restrict__ rpb, const int* __restrict__ relidx,
    float* __restrict__ out) {
    extern __shared__ unsigned short smem[];
    unsigned short* XB = smem + XB_OFF;   // [64][XS]  x (bf16), later attn_out
    unsigned short* QK = smem + QK_OFF;   // [64][QKS] q*scale | k
    unsigned short* VT = smem + VT_OFF;   // [256][VTS] v transposed (dim-major)
    unsigned short* PB = smem + PB_OFF;   // [8 waves][16][PBS] softmax probs

    const int b    = blockIdx.x;
    const int tid  = threadIdx.x;
    // wave id is wave-invariant: force it into an SGPR so tile routing is scalar
    const int w    = __builtin_amdgcn_readfirstlane(tid >> 5);
    const int lane = tid & 31;
    const int half = lane >> 4;
    const int lr   = lane & 15;

    // ---------- Phase 1: stage x[b] (fp32) -> bf16 LDS ----------
    const float* xb = x + (size_t)b * NTOK * DIM;
    for (int j = 0; j < 16; ++j) {
        int c   = tid + j * 256;            // float4 chunk id, 4096 total
        int row = c >> 6;
        int col = (c & 63) << 2;
        float4 f = ((const float4*)xb)[c];
        unsigned int lo = (unsigned int)f2bf(f.x) | ((unsigned int)f2bf(f.y) << 16);
        unsigned int hi = (unsigned int)f2bf(f.z) | ((unsigned int)f2bf(f.w) << 16);
        *(uint2*)(XB + row * XS + col) = make_uint2(lo, hi);
    }
    __syncthreads();

    // ---------- Phase 2: qkv[64,768] = x @ Wqkv + bias; q pre-scaled ----------
    const float SCALE = 0.17677669529663687f;  // 32^-0.5
    for (int t = 0; t < 24; ++t) {
        int T  = w + 8 * t;                    // 192 tiles, disjoint across waves
        int mt = T & 3;                        // scalar
        int nt = T >> 2;                       // scalar
        const unsigned short* brow = wt + (size_t)(nt * 16) * DIM;
        __builtin_prefetch(brow + lr * DIM, 0, 1);
        f32x8 acc = {};
        for (int kb = 0; kb < 8; ++kb) {
            bf16x16 a  = load_a16(XB + mt * 16 * XS, XS, kb * 32, lane);
            bf16x16 bb = load_b16(brow, DIM, kb * 32, lane);
            acc = wmma_bf16(a, bb, acc);
        }
        // q/k/v routing is wave-uniform in nt (lr < 16 => c-range within region)
        int c       = nt * 16 + lr;
        int rowbase = mt * 16 + 8 * half;
        if (nt < 16) {                         // ---- q: scale + bias ----
            float bias = bq[c];
            #pragma unroll
            for (int r = 0; r < 8; ++r)
                QK[(rowbase + r) * QKS + c] = f2bf((acc[r] + bias) * SCALE);
        } else if (nt < 32) {                  // ---- k ----
            float bias = bkv[c - DIM];
            #pragma unroll
            for (int r = 0; r < 8; ++r)
                QK[(rowbase + r) * QKS + c] = f2bf(acc[r] + bias);
        } else {                               // ---- v: store transposed ----
            float bias = bkv[c - DIM];
            #pragma unroll
            for (int r = 0; r < 8; ++r)
                VT[(c - 2 * DIM) * VTS + rowbase + r] = f2bf(acc[r] + bias);
        }
    }
    __syncthreads();

    // ---------- Phase 3: per-head attention (wave w == head h) ----------
    const int h = w;                            // scalar
    unsigned short* Pw = PB + w * 16 * PBS;
    for (int mt = 0; mt < 4; ++mt) {
        // attn tile row-block: q[mt] @ k^T  (K = 32, one WMMA per n-tile)
        bf16x16 qa = load_a16(QK + mt * 16 * QKS, QKS, h * HEAD_DIM, lane);
        f32x8 acc[4];
        #pragma unroll
        for (int nt = 0; nt < 4; ++nt) {
            bf16x16 kb = load_b16(QK + nt * 16 * QKS + DIM + h * HEAD_DIM,
                                  QKS, 0, lane);
            f32x8 z = {};
            acc[nt] = wmma_bf16(qa, kb, z);
        }
        // bias + softmax across the 64-wide row (spread over 16 lanes of a half)
        #pragma unroll
        for (int r = 0; r < 8; ++r) {
            int M = mt * 16 + r + 8 * half;
            float m = -1e30f;
            #pragma unroll
            for (int nt = 0; nt < 4; ++nt) {
                int N = nt * 16 + lr;
                float v = acc[nt][r] + rpb[relidx[M * 64 + N] * HEADS + h];
                acc[nt][r] = v;
                m = fmaxf(m, v);
            }
            m = fmaxf(m, __shfl_xor(m, 1, 32));
            m = fmaxf(m, __shfl_xor(m, 2, 32));
            m = fmaxf(m, __shfl_xor(m, 4, 32));
            m = fmaxf(m, __shfl_xor(m, 8, 32));
            float s = 0.f;
            #pragma unroll
            for (int nt = 0; nt < 4; ++nt) {
                float e = __expf(acc[nt][r] - m);
                acc[nt][r] = e;
                s += e;
            }
            s += __shfl_xor(s, 1, 32);
            s += __shfl_xor(s, 2, 32);
            s += __shfl_xor(s, 4, 32);
            s += __shfl_xor(s, 8, 32);
            float inv = 1.0f / s;
            #pragma unroll
            for (int nt = 0; nt < 4; ++nt)
                Pw[(r + 8 * half) * PBS + nt * 16 + lr] = f2bf(acc[nt][r] * inv);
        }
        __syncthreads();   // uniform: publish P before fragment reload

        // out rows [mt]: P[16,64] @ v[64,32]  (K=64 -> 2 WMMA per nd tile)
        #pragma unroll
        for (int nd = 0; nd < 2; ++nd) {
            f32x8 oacc = {};
            #pragma unroll
            for (int kb = 0; kb < 2; ++kb) {
                bf16x16 pa = load_a16(Pw, PBS, kb * 32, lane);
                bf16x16 vb = load_b16(VT + (h * HEAD_DIM + nd * 16) * VTS,
                                      VTS, kb * 32, lane);
                oacc = wmma_bf16(pa, vb, oacc);
            }
            #pragma unroll
            for (int r = 0; r < 8; ++r) {
                int row = mt * 16 + r + 8 * half;
                XB[row * XS + h * HEAD_DIM + nd * 16 + lr] = f2bf(oacc[r]);
            }
        }
        __syncthreads();   // uniform: Pw reused next mt iteration
    }
    __syncthreads();

    // ---------- Phase 4: y[64,256] = attn_out @ wp + bp ----------
    for (int t = 0; t < 8; ++t) {
        int T  = w + 8 * t;                    // 64 tiles, disjoint across waves
        int mt = T & 3;                        // scalar
        int nt = T >> 2;                       // scalar
        const unsigned short* brow = wpt + (size_t)(nt * 16) * DIM;
        __builtin_prefetch(brow + lr * DIM, 0, 1);
        f32x8 acc = {};
        for (int kb = 0; kb < 8; ++kb) {
            bf16x16 a  = load_a16(XB + mt * 16 * XS, XS, kb * 32, lane);
            bf16x16 bb = load_b16(brow, DIM, kb * 32, lane);
            acc = wmma_bf16(a, bb, acc);
        }
        int c = nt * 16 + lr;
        float bias = bp[c];
        float* orow = out + ((size_t)b * NTOK + mt * 16 + 8 * half) * DIM + c;
        #pragma unroll
        for (int r = 0; r < 8; ++r)
            orow[(size_t)r * DIM] = acc[r] + bias;
    }
}

extern "C" void kernel_launch(void* const* d_in, const int* in_sizes, int n_in,
                              void* d_out, int out_size, void* d_ws, size_t ws_size,
                              hipStream_t stream) {
    const float* x    = (const float*)d_in[0];
    const float* wq   = (const float*)d_in[1];
    const float* bq   = (const float*)d_in[2];
    const float* wkv  = (const float*)d_in[3];
    const float* bkv  = (const float*)d_in[4];
    const float* wp   = (const float*)d_in[5];
    const float* bp   = (const float*)d_in[6];
    const float* rpb  = (const float*)d_in[7];
    const int*   ridx = (const int*)d_in[8];
    float* out = (float*)d_out;

    unsigned short* wt  = (unsigned short*)d_ws;          // 768*256 bf16
    unsigned short* wpt = wt + QKV_DIM * DIM;             // 256*256 bf16

    // one-time bf16 transpose of all weights (L2-resident afterwards)
    convert_weights_kernel<<<(QKV_DIM * DIM + DIM * DIM) / 256, 256, 0, stream>>>(
        wq, wkv, wp, wt, wpt);

    size_t smem = (size_t)SMEM_ELEMS * sizeof(unsigned short);  // ~152 KB
    TotalAttention_kernel<<<4096, 256, smem, stream>>>(
        x, wt, bq, bkv, wpt, bp, rpb, ridx, out);
}
